// ALIGNN_52106543235179
// MI455X (gfx1250) — matrix operand order repaired and checked by reading, not compiled, not run on hardware
//
#include <hip/hip_runtime.h>
#include <hip/hip_bf16.h>

typedef __attribute__((ext_vector_type(16))) _Float16 v16h;
typedef __attribute__((ext_vector_type(8)))  float    v8f;

#define Nn 100000
#define Ee 800000
#define Hh 96
#define NH (Nn * Hh)
#define EH (Ee * Hh)

// ---------------------------------------------------------------------------
// WMMA fragment helpers (v_wmma_f32_16x16x32_f16, wave32)
// A: 16x32 f16. lanes 0-15: M=lane, v0..3 K=0..7, v4..7 K=16..23
//               lanes16-31: M=lane-16, K offset +8
// B: 32x16 f16. N = lane & 15, 16 halves per lane = K (same K order as A)
// C: 8 x f32.   M = i + 8*(lane>>4), N = lane & 15
// ---------------------------------------------------------------------------
__device__ inline v16h load_a_frag(const _Float16* As, int lane, int kk) {
    union { v16h v; unsigned u[8]; } a;
    const int M  = lane & 15;
    const int hi = (lane & 16) ? 8 : 0;
    const _Float16* row = As + M * Hh + kk * 32 + hi;
#pragma unroll
    for (int v = 0; v < 8; ++v) {
        const int k0 = (v < 4) ? (2 * v) : (16 + 2 * (v - 4));
        a.u[v] = *reinterpret_cast<const unsigned*>(row + k0);
    }
    return a.v;
}

__device__ inline v16h load_b_frag(const _Float16* W, int lane, int kk, int nt) {
    // W row-major [96][96], f16; N col = nt*16+(lane&15), K = kk*32 + (s<8 ? hi+s : 16+hi+s-8), hi = 8*(lane>>4)
    v16h b;
    const int n  = nt * 16 + (lane & 15);
    const int hi = (lane & 16) ? 8 : 0;
#pragma unroll
    for (int s = 0; s < 16; ++s) {
        const int k = (s < 8) ? (hi + s) : (16 + hi + (s - 8));
        b[s] = W[(kk * 32 + k) * Hh + n];
    }
    return b;
}

__device__ inline v8f wmma_f16(v16h a, v16h b, v8f c) {
    return __builtin_amdgcn_wmma_f32_16x16x32_f16(false, a, false, b, (short)0, c,
                                                  false, false);
}

// Force the native fire-and-forget global_atomic_add_f32 (no CAS loop):
// accumulators live in coarse-grained device workspace, so this is safe.
__device__ inline void atomic_add_f32(float* p, float v) {
#if defined(__HIP_DEVICE_COMPILE__)
    unsafeAtomicAdd(p, v);
#else
    atomicAdd(p, v);
#endif
}

// ---------------------------------------------------------------------------
// Kernel 0: zero scatter accumulators + convert 5 weight matrices to f16
// ---------------------------------------------------------------------------
__global__ void alignn_prep_kernel(const float* __restrict__ Wsg,
                                   const float* __restrict__ Wdg,
                                   const float* __restrict__ Weg,
                                   const float* __restrict__ Wdu,
                                   const float* __restrict__ Wsu,
                                   _Float16* __restrict__ Wf16,
                                   float* __restrict__ ssh,
                                   float* __restrict__ ss) {
    const int stride = gridDim.x * blockDim.x;
    for (int i = blockIdx.x * blockDim.x + threadIdx.x; i < NH; i += stride) {
        ssh[i] = 0.0f;
        ss[i]  = 0.0f;
    }
    for (int i = blockIdx.x * blockDim.x + threadIdx.x; i < 5 * Hh * Hh; i += stride) {
        const int mi = i / (Hh * Hh);
        const int j  = i % (Hh * Hh);
        const float* W = (mi == 0) ? Wsg : (mi == 1) ? Wdg : (mi == 2) ? Weg
                        : (mi == 3) ? Wdu : Wsu;
        Wf16[i] = (_Float16)W[j];
    }
}

// ---------------------------------------------------------------------------
// Kernel 1: tp = time_feats @ W_tp + b_tp   (1 x H, trivial)
// ---------------------------------------------------------------------------
__global__ void alignn_tp_kernel(const float* __restrict__ timef,
                                 const float* __restrict__ Wtp,
                                 const float* __restrict__ btp,
                                 float* __restrict__ tp) {
    const int n = threadIdx.x;
    if (n < Hh) {
        float acc = btp[n];
        for (int k = 0; k < Hh; ++k) acc += timef[k] * Wtp[k * Hh + n];
        tp[n] = acc;
    }
}

// ---------------------------------------------------------------------------
// Kernel 2: fused node-side GEMMs (4 weight matrices) via WMMA.
// Block = 192 threads (6 waves), one 16-row node tile per block.
// ---------------------------------------------------------------------------
__global__ void __launch_bounds__(192)
alignn_node_gemm_kernel(const float* __restrict__ nodef,
                        const _Float16* __restrict__ Wf16,
                        const float* __restrict__ bsg,
                        const float* __restrict__ bdg,
                        const float* __restrict__ bdu,
                        const float* __restrict__ bsu,
                        const float* __restrict__ tp,
                        float* __restrict__ e_src, float* __restrict__ e_dst,
                        float* __restrict__ Bh, float* __restrict__ xs) {
    __shared__ _Float16 As[16 * Hh];

    const int  t    = threadIdx.x;
    const long row0 = (long)blockIdx.x * 16;
    const float* srcp = nodef + row0 * Hh;
#pragma unroll
    for (int j = 0; j < 8; ++j) {
        const int idx = t * 8 + j;
        As[idx] = (_Float16)srcp[idx];
    }
    __syncthreads();

    const int lane = t & 31;
    const int nt   = t >> 5;  // N-tile 0..5

    v8f c0 = {}, c1 = {}, c2 = {}, c3 = {};
#pragma unroll
    for (int kk = 0; kk < 3; ++kk) {
        const v16h a  = load_a_frag(As, lane, kk);
        const v16h b0 = load_b_frag(Wf16 + 0 * Hh * Hh, lane, kk, nt);  // W_sg
        const v16h b1 = load_b_frag(Wf16 + 1 * Hh * Hh, lane, kk, nt);  // W_dg
        const v16h b2 = load_b_frag(Wf16 + 3 * Hh * Hh, lane, kk, nt);  // W_du
        const v16h b3 = load_b_frag(Wf16 + 4 * Hh * Hh, lane, kk, nt);  // W_su
        c0 = wmma_f16(a, b0, c0);
        c1 = wmma_f16(a, b1, c1);
        c2 = wmma_f16(a, b2, c2);
        c3 = wmma_f16(a, b3, c3);
    }

    const int Ncol = lane & 15;
    const int Mb   = (lane >> 4) << 3;
    const int col  = nt * 16 + Ncol;
    const float v_sg = bsg[col] + tp[col];
    const float v_dg = bdg[col];
    const float v_du = bdu[col];
    const float v_su = bsu[col];
#pragma unroll
    for (int i = 0; i < 8; ++i) {
        const long o = (row0 + Mb + i) * Hh + col;
        e_src[o] = c0[i] + v_sg;
        e_dst[o] = c1[i] + v_dg;
        Bh[o]    = c2[i] + v_du;
        xs[o]    = c3[i] + v_su;
    }
}

// ---------------------------------------------------------------------------
// Kernel 3: edge GEMM (WMMA) + gather + sigmoid + scatter atomics +
//           LayerNorm/SiLU/residual producing y directly (m never hits HBM).
// Block = 192 threads, one 16-edge tile per block.
// ---------------------------------------------------------------------------
__global__ void __launch_bounds__(192)
alignn_edge_kernel(const float* __restrict__ edgef,
                   const int* __restrict__ srcI, const int* __restrict__ dstI,
                   const _Float16* __restrict__ Weg16,
                   const float* __restrict__ beg,
                   const float* __restrict__ e_src, const float* __restrict__ e_dst,
                   const float* __restrict__ Bh,
                   const float* __restrict__ g_ln_e, const float* __restrict__ b_ln_e,
                   float* __restrict__ ssh, float* __restrict__ ss,
                   float* __restrict__ y_out) {
    __shared__ _Float16 As[16 * Hh];
    __shared__ float    Ef[16 * Hh];
    __shared__ float    Ms[16][Hh + 1];
    __shared__ int      sS[16], sD[16];
    __shared__ float    sMu[16], sRs[16];

    const int  t  = threadIdx.x;
    const long e0 = (long)blockIdx.x * 16;
    const float* ep = edgef + e0 * Hh;
#pragma unroll
    for (int j = 0; j < 8; ++j) {
        const int idx = t * 8 + j;
        const float v = ep[idx];
        Ef[idx] = v;
        As[idx] = (_Float16)v;
    }
    if (t < 16) {
        sS[t] = srcI[e0 + t];
        sD[t] = dstI[e0 + t];
    }
    __syncthreads();

    const int lane = t & 31;
    const int nt   = t >> 5;

    v8f c = {};
#pragma unroll
    for (int kk = 0; kk < 3; ++kk) {
        const v16h a = load_a_frag(As, lane, kk);
        const v16h b = load_b_frag(Weg16, lane, kk, nt);
        c = wmma_f16(a, b, c);
    }

    const int Ncol = lane & 15;
    const int Mb   = (lane >> 4) << 3;
    const int col  = nt * 16 + Ncol;
    const float bb = beg[col];
#pragma unroll
    for (int i = 0; i < 8; ++i) {
        const int  r = Mb + i;
        const long s = sS[r];
        const long d = sD[r];
        const float m   = c[i] + bb + e_src[s * Hh + col] + e_dst[d * Hh + col];
        const float sig = 1.0f / (1.0f + __expf(-m));
        atomic_add_f32(&ssh[d * Hh + col], sig * Bh[s * Hh + col]);
        atomic_add_f32(&ss[d * Hh + col], sig);
        Ms[r][col] = m;
    }
    __syncthreads();

    if (t < 16) {
        float sum = 0.0f, sq = 0.0f;
        for (int k = 0; k < Hh; ++k) {
            const float v = Ms[t][k];
            sum += v;
            sq  += v * v;
        }
        const float mu  = sum * (1.0f / Hh);
        const float var = sq * (1.0f / Hh) - mu * mu;
        sMu[t] = mu;
        sRs[t] = rsqrtf(var + 1e-5f);
    }
    __syncthreads();

#pragma unroll
    for (int j = 0; j < 8; ++j) {
        const int idx = t * 8 + j;
        const int r   = idx / Hh;
        const int cc  = idx % Hh;
        const float z  = (Ms[r][cc] - sMu[r]) * sRs[r] * g_ln_e[cc] + b_ln_e[cc];
        const float sl = z / (1.0f + __expf(-z));
        y_out[e0 * Hh + idx] = Ef[idx] + sl;
    }
}

// ---------------------------------------------------------------------------
// Kernel 4: node finalize: x = node + silu(LN(xs + ssh/(ss+1e-6)))
// One wave per node row (3 cols/lane), wave32 shuffle reduction.
// ---------------------------------------------------------------------------
__global__ void __launch_bounds__(256)
alignn_node_out_kernel(const float* __restrict__ nodef,
                       const float* __restrict__ xs,
                       const float* __restrict__ ssh, const float* __restrict__ ss,
                       const float* __restrict__ g_ln_n, const float* __restrict__ b_ln_n,
                       float* __restrict__ x_out) {
    const int  lane = threadIdx.x & 31;
    const long r    = (long)blockIdx.x * 8 + (threadIdx.x >> 5);
    const long base = r * Hh;

    float x[3];
    float s = 0.0f, q = 0.0f;
#pragma unroll
    for (int j = 0; j < 3; ++j) {
        const int cc = lane + j * 32;
        const float h = ssh[base + cc] / (ss[base + cc] + 1e-6f);
        const float v = xs[base + cc] + h;
        x[j] = v;
        s += v;
        q += v * v;
    }
#pragma unroll
    for (int off = 16; off > 0; off >>= 1) {
        s += __shfl_xor(s, off, 32);
        q += __shfl_xor(q, off, 32);
    }
    const float mu   = s * (1.0f / Hh);
    const float rstd = rsqrtf(q * (1.0f / Hh) - mu * mu + 1e-5f);
#pragma unroll
    for (int j = 0; j < 3; ++j) {
        const int cc = lane + j * 32;
        const float z  = (x[j] - mu) * rstd * g_ln_n[cc] + b_ln_n[cc];
        const float sl = z / (1.0f + __expf(-z));
        x_out[base + cc] = nodef[base + cc] + sl;
    }
}

// ---------------------------------------------------------------------------
// Launcher
// ---------------------------------------------------------------------------
extern "C" void kernel_launch(void* const* d_in, const int* in_sizes, int n_in,
                              void* d_out, int out_size, void* d_ws, size_t ws_size,
                              hipStream_t stream) {
    (void)in_sizes; (void)n_in; (void)out_size; (void)ws_size;

    const float* nodef = (const float*)d_in[0];
    const float* edgef = (const float*)d_in[1];
    const float* timef = (const float*)d_in[2];
    const int*   srcI  = (const int*)d_in[3];
    const int*   dstI  = (const int*)d_in[4];
    const float* W_tp = (const float*)d_in[5];
    const float* b_tp = (const float*)d_in[6];
    const float* W_sg = (const float*)d_in[7];
    const float* b_sg = (const float*)d_in[8];
    const float* W_dg = (const float*)d_in[9];
    const float* b_dg = (const float*)d_in[10];
    const float* W_eg = (const float*)d_in[11];
    const float* b_eg = (const float*)d_in[12];
    const float* W_su = (const float*)d_in[13];
    const float* b_su = (const float*)d_in[14];
    const float* W_du = (const float*)d_in[15];
    const float* b_du = (const float*)d_in[16];
    const float* g_ln_e = (const float*)d_in[17];
    const float* b_ln_e = (const float*)d_in[18];
    const float* g_ln_n = (const float*)d_in[19];
    const float* b_ln_n = (const float*)d_in[20];

    // Workspace layout (256B-aligned regions)
    char* ws = (char*)d_ws;
    _Float16* Wf16 = (_Float16*)ws;                 // 5 * 96*96 halves = 92160 B
    float* tp    = (float*)(ws + 92160);            // 384 B (+pad)
    float* e_src = (float*)(ws + 92672);            // NH floats
    float* e_dst = e_src + NH;
    float* Bh    = e_dst + NH;
    float* xs    = Bh + NH;
    float* ssh   = xs + NH;
    float* ss    = ssh + NH;

    float* x_out = (float*)d_out;       // [N,H]
    float* y_out = x_out + NH;          // [E,H]

    // 0) zero accumulators + weight f16 conversion (slots: sg,dg,eg,du,su)
    alignn_prep_kernel<<<2048, 256, 0, stream>>>(W_sg, W_dg, W_eg, W_du, W_su,
                                                 Wf16, ssh, ss);
    // 1) tp row
    alignn_tp_kernel<<<1, 128, 0, stream>>>(timef, W_tp, b_tp, tp);
    // 2) node GEMMs (N/16 = 6250 tiles)
    alignn_node_gemm_kernel<<<Nn / 16, 192, 0, stream>>>(
        nodef, Wf16, b_sg, b_dg, b_du, b_su, tp, e_src, e_dst, Bh, xs);
    // 3) edge GEMM + gather/scatter + LN/SiLU -> y (E/16 = 50000 tiles)
    alignn_edge_kernel<<<Ee / 16, 192, 0, stream>>>(
        edgef, srcI, dstI, Wf16 + 2 * Hh * Hh, b_eg, e_src, e_dst, Bh,
        g_ln_e, b_ln_e, ssh, ss, y_out);
    // 4) node finalize -> x (N/8 = 12500 blocks)
    alignn_node_out_kernel<<<Nn / 8, 256, 0, stream>>>(
        nodef, xs, ssh, ss, g_ln_n, b_ln_n, x_out);
}
